// VisionAttention_9405978378411
// MI455X (gfx1250) — matrix operand order repaired and checked by reading, not compiled, not run on hardware
//
#include <hip/hip_runtime.h>

// ---------------------------------------------------------------------------
// VisionAttention forward for MI455X (gfx1250, wave32, WMMA, async-to-LDS).
//
// Roofline: ~81 GFLOP total, ~100 MB of unavoidable HBM traffic -> memory
// bound at 23.3 TB/s (~5us floor). Flash-attention structure (no [B,H,S,S]
// materialization, which would add ~1.6 GB of traffic). bf16 WMMA
// (v_wmma_f32_16x16x32_bf16, fp32 accum) is the high-throughput matrix path.
// K/V tiles are moved HBM->LDS with GLOBAL_LOAD_ASYNC_TO_LDS_B128 (ASYNCcnt),
// double-buffered so the next tile's traffic overlaps the current tile's
// WMMAs; V is stored head-transposed by the GEMM so no LDS transpose needed.
// GEMM LDS tile fills use packed ds_store_b128; P staging uses scalar b16 stores.
// ---------------------------------------------------------------------------

#define BDIM 4
#define SDIM 2048
#define DDIM 768
#define HDIM 12
#define HD   64
#define QK_SCALE 0.125f            // HD^-0.5

typedef __attribute__((ext_vector_type(16))) __bf16 v16bf;
typedef __attribute__((ext_vector_type(8)))  __bf16 v8bf;
typedef __attribute__((ext_vector_type(8)))  float  v8f;

union AFrag { v16bf v; v8bf h[2]; };

__device__ __forceinline__ unsigned lds_off(const void* p) {
    // generic LDS pointers carry the LDS byte offset in the low 32 bits
    return (unsigned)(size_t)p;
}

// ===========================================================================
// Kernel 1: QKV projection.  qkv[M=8192, N=2304] = x[8192,768] @ w[768,2304]+b
// Output scattered to head-major bf16 workspaces: Q,K as [bh][s][hd] (Q
// pre-scaled by HD^-0.5), V transposed as [bh][hd][s] so the attention
// kernel's V tiles are direct 2-D copies.
// Block tile 128x64, BK=32, 256 threads = 8 waves (4x2), wave tile 32x32.
// ===========================================================================
__global__ __launch_bounds__(256)
void qkv_gemm_kernel(const float* __restrict__ x,
                     const float* __restrict__ w,
                     const float* __restrict__ bias,
                     __bf16* __restrict__ qws,
                     __bf16* __restrict__ kws,
                     __bf16* __restrict__ vws)
{
    constexpr int BM = 128, BN = 64, BK = 32;
    constexpr int LDA = 40;   // bf16 elems; 80B rows -> 16B-aligned frag loads
    constexpr int LDB = 40;
    __shared__ __align__(16) __bf16 sa[BM * LDA];   // A tile, [m][k]
    __shared__ __align__(16) __bf16 sbt[BN * LDB];  // B tile transposed, [n][k]

    const int tid  = threadIdx.x;
    const int lane = tid & 31;
    const int wave = tid >> 5;
    const int wm   = (wave & 3) * 32;          // wave M offset in block tile
    const int wn   = (wave >> 2) * 32;         // wave N offset
    const int bm   = (blockIdx.x % 64) * BM;   // 64 M tiles
    const int bn   = (blockIdx.x / 64) * BN;   // 36 N tiles

    const int rlo = lane & 15;
    const int kbA = (lane < 16) ? 0 : 8;       // A-frag K base (ISA layout)
    const int kbB = (lane < 16) ? 0 : 8;       // B-frag K base

    // B-tile loader thread mapping: one n column, 8 consecutive k per thread
    const int nl = tid & 63;                   // n within tile (0..63)
    const int kg = (tid >> 6) << 3;            // k group base (0,8,16,24)

    v8f acc[2][2] = {};

    for (int k0 = 0; k0 < DDIM; k0 += BK) {
        // --- A tile: 128x32 fp32 -> bf16; 2 float4 -> one ds_store_b128 ---
        #pragma unroll
        for (int i = 0; i < 2; ++i) {
            int idx = tid + i * 256;                 // 0..511
            int r = idx >> 2, c8 = (idx & 3) << 3;   // row, 8-elem k chunk
            const float* xp = x + (size_t)(bm + r) * DDIM + k0 + c8;
            const float4 f0 = *(const float4*)(xp);
            const float4 f1 = *(const float4*)(xp + 4);
            v8bf pk;
            pk[0] = (__bf16)f0.x; pk[1] = (__bf16)f0.y;
            pk[2] = (__bf16)f0.z; pk[3] = (__bf16)f0.w;
            pk[4] = (__bf16)f1.x; pk[5] = (__bf16)f1.y;
            pk[6] = (__bf16)f1.z; pk[7] = (__bf16)f1.w;
            *(v8bf*)(sa + r * LDA + c8) = pk;
        }
        // --- B tile 32(k)x64(n): per-thread column gather (each dword load
        //     is wave-coalesced across n), pack 8 k -> one ds_store_b128
        //     into the transposed [n][k] tile ---
        {
            const float* wp = w + (size_t)(k0 + kg) * (3 * DDIM) + bn + nl;
            v8bf pk;
            #pragma unroll
            for (int j = 0; j < 8; ++j)
                pk[j] = (__bf16)wp[(size_t)j * (3 * DDIM)];
            *(v8bf*)(sbt + nl * LDB + kg) = pk;
        }
        __syncthreads();

        #pragma unroll
        for (int mt = 0; mt < 2; ++mt) {
            AFrag a;
            const __bf16* ap = sa + (wm + mt * 16 + rlo) * LDA + kbA;
            a.h[0] = *(const v8bf*)(ap);        // K = kbA .. kbA+7
            a.h[1] = *(const v8bf*)(ap + 16);   // K = kbA+16 .. kbA+23
            #pragma unroll
            for (int nt = 0; nt < 2; ++nt) {
                AFrag b;
                const __bf16* bp = sbt + (wn + nt * 16 + rlo) * LDB + kbB;
                b.h[0] = *(const v8bf*)(bp);       // K = kbB .. kbB+7
                b.h[1] = *(const v8bf*)(bp + 16);  // K = kbB+16 .. kbB+23
                acc[mt][nt] = __builtin_amdgcn_wmma_f32_16x16x32_bf16(
                    false, a.v, false, b.v, (short)0, acc[mt][nt], false, false);
            }
        }
        __syncthreads();
    }

    // --- epilogue: bias, Q scaling, scatter to head-major bf16 Q/K/V ---
    const int rhi = (lane >> 4) << 3;   // +8 rows for upper half-wave
    #pragma unroll
    for (int mt = 0; mt < 2; ++mt)
    #pragma unroll
    for (int nt = 0; nt < 2; ++nt)
    #pragma unroll
    for (int r = 0; r < 8; ++r) {
        int gm = bm + wm + mt * 16 + r + rhi;       // token index (0..8191)
        int gn = bn + wn + nt * 16 + rlo;           // qkv column (0..2303)
        float v = acc[mt][nt][r] + bias[gn];
        int which = gn / DDIM;                      // 0=q 1=k 2=v
        int rem   = gn % DDIM;
        int h     = rem >> 6;
        int hd    = rem & 63;
        int b     = gm >> 11;                       // / 2048
        int s     = gm & 2047;
        int bh    = b * HDIM + h;
        size_t o;
        if (which == 2)
            o = ((size_t)bh * HD + hd) * SDIM + s;        // V transposed
        else
            o = (((size_t)bh * SDIM + s) << 6) + hd;      // Q,K row-major
        if (which == 0) v *= QK_SCALE;
        __bf16* dst = (which == 0) ? qws : (which == 1) ? kws : vws;
        dst[o] = (__bf16)v;
    }
}

// ===========================================================================
// Kernel 2: flash attention.  Grid: (S/64 q-tiles, B*H heads), 128 threads.
// Each wave owns 16 q rows; iterate 64-key tiles with online softmax.
// K/V tiles double-buffered in LDS, filled by GLOBAL_LOAD_ASYNC_TO_LDS_B128.
// ===========================================================================
__global__ __launch_bounds__(128)
void flash_attn_kernel(const __bf16* __restrict__ qws,
                       const __bf16* __restrict__ kws,
                       const __bf16* __restrict__ vws,
                       float* __restrict__ out)
{
    constexpr int LDK = 72;                          // 144B rows (16B aligned)
    __shared__ __align__(16) __bf16 sk[2][64 * LDK]; // K tiles [key][d]
    __shared__ __align__(16) __bf16 sv[2][64 * LDK]; // V tiles [d][key]
    __shared__ __align__(16) __bf16 sp[4 * 16 * LDK];// per-wave P staging

    const int tid  = threadIdx.x;
    const int lane = tid & 31;
    const int wave = tid >> 5;
    const int qt   = blockIdx.x;            // q tile (0..31)
    const int bh   = blockIdx.y;            // head-batch (0..47)
    const int b    = bh / HDIM;
    const int h    = bh % HDIM;

    const __bf16* qhead  = qws + (size_t)bh * SDIM * HD;
    const __bf16* khead  = kws + (size_t)bh * SDIM * HD;
    const __bf16* vthead = vws + (size_t)bh * HD * SDIM;  // [hd][s]

    const int rlo = lane & 15;
    const int rhi = (lane >> 4) << 3;
    const int kbA = (lane < 16) ? 0 : 8;
    const int kbB = (lane < 16) ? 0 : 8;
    const int qrow0 = qt * 64 + wave * 16;

    // Async HBM->LDS copy of one 64-key tile of K and V (16B chunks,
    // 4 K-chunks + 4 V-chunks per thread). No VGPR staging, ASYNCcnt-tracked.
    auto issue_tile = [&](int buf, int kt) {
        #pragma unroll
        for (int i = 0; i < 4; ++i) {
            int c   = tid + i * 128;                // 0..511 chunk id
            int row = c >> 3;                       // 0..63
            int col = (c & 7) << 3;                 // element offset, 16B units
            const __bf16* kg = khead + (size_t)(kt + row) * HD + col;
            unsigned kl = lds_off(&sk[buf][row * LDK + col]);
            asm volatile("global_load_async_to_lds_b128 %0, %1, off"
                         :: "v"(kl), "v"(kg) : "memory");
            const __bf16* vg = vthead + (size_t)row * SDIM + kt + col;
            unsigned vl = lds_off(&sv[buf][row * LDK + col]);
            asm volatile("global_load_async_to_lds_b128 %0, %1, off"
                         :: "v"(vl), "v"(vg) : "memory");
        }
    };

    // Q fragments: rows qrow0+rlo, d-chunks 0 and 32 (Q already scaled)
    AFrag qa[2];
    #pragma unroll
    for (int c = 0; c < 2; ++c) {
        const __bf16* qp = qhead + (size_t)(qrow0 + rlo) * HD + c * 32 + kbA;
        qa[c].h[0] = *(const v8bf*)(qp);
        qa[c].h[1] = *(const v8bf*)(qp + 16);
    }

    float m[8], l[8];
    v8f o[4] = {};
    #pragma unroll
    for (int r = 0; r < 8; ++r) { m[r] = -3.0e38f; l[r] = 0.0f; }

    issue_tile(0, 0);

    for (int kt = 0; kt < SDIM; kt += 64) {
        const int cur = (kt >> 6) & 1;
        // my async copies (tile `cur`) done; barrier also proves every wave
        // finished reading buffer cur^1 last iteration, so it can be refilled.
        asm volatile("s_wait_asynccnt 0x0" ::: "memory");
        __syncthreads();
        if (kt + 64 < SDIM) issue_tile(cur ^ 1, kt + 64);

        // --- scores S = Q K^T for 4 subtiles of 16 keys ---
        float s[4][8];
        #pragma unroll
        for (int nt = 0; nt < 4; ++nt) {
            v8f c = {};
            #pragma unroll
            for (int dc = 0; dc < 2; ++dc) {
                AFrag bf;
                const __bf16* bp = &sk[cur][(nt * 16 + rlo) * LDK + dc * 32 + kbB];
                bf.h[0] = *(const v8bf*)(bp);
                bf.h[1] = *(const v8bf*)(bp + 16);
                c = __builtin_amdgcn_wmma_f32_16x16x32_bf16(
                    false, qa[dc].v, false, bf.v, (short)0, c, false, false);
            }
            #pragma unroll
            for (int r = 0; r < 8; ++r) s[nt][r] = c[r];
        }

        // --- online softmax + stage P in per-wave LDS (C-layout -> A-layout;
        //     same-wave DS ordering is enforced by the compiler's dscnt waits)
        #pragma unroll
        for (int r = 0; r < 8; ++r) {
            float rowmax = fmaxf(fmaxf(s[0][r], s[1][r]), fmaxf(s[2][r], s[3][r]));
            #pragma unroll
            for (int mk = 8; mk >= 1; mk >>= 1)
                rowmax = fmaxf(rowmax, __shfl_xor(rowmax, mk, 32));
            float mnew  = fmaxf(m[r], rowmax);
            float alpha = __expf(m[r] - mnew);
            m[r] = mnew;
            float psum = 0.0f;
            #pragma unroll
            for (int nt = 0; nt < 4; ++nt) {
                s[nt][r] = __expf(s[nt][r] - mnew);
                psum += s[nt][r];
            }
            #pragma unroll
            for (int mk = 8; mk >= 1; mk >>= 1)
                psum += __shfl_xor(psum, mk, 32);
            l[r] = l[r] * alpha + psum;
            #pragma unroll
            for (int nt = 0; nt < 4; ++nt) o[nt][r] *= alpha;

            __bf16* pp = sp + (wave * 16 + r + rhi) * LDK;
            #pragma unroll
            for (int nt = 0; nt < 4; ++nt)
                pp[nt * 16 + rlo] = (__bf16)s[nt][r];
        }

        // --- O += P @ V ---
        #pragma unroll
        for (int nt = 0; nt < 4; ++nt) {          // d tiles of 16
            #pragma unroll
            for (int kc = 0; kc < 2; ++kc) {      // key chunks of 32
                AFrag pa, vb;
                const __bf16* pp = sp + (wave * 16 + rlo) * LDK + kc * 32 + kbA;
                pa.h[0] = *(const v8bf*)(pp);
                pa.h[1] = *(const v8bf*)(pp + 16);
                const __bf16* vp = &sv[cur][(nt * 16 + rlo) * LDK + kc * 32 + kbB];
                vb.h[0] = *(const v8bf*)(vp);
                vb.h[1] = *(const v8bf*)(vp + 16);
                o[nt] = __builtin_amdgcn_wmma_f32_16x16x32_bf16(
                    false, pa.v, false, vb.v, (short)0, o[nt], false, false);
            }
        }
    }

    // --- normalize and write out [B,S,D] fp32 ---
    #pragma unroll
    for (int nt = 0; nt < 4; ++nt)
    #pragma unroll
    for (int r = 0; r < 8; ++r) {
        int srow = qrow0 + r + rhi;
        int d    = h * HD + nt * 16 + rlo;
        out[(size_t)(b * SDIM + srow) * DDIM + d] = o[nt][r] / l[r];
    }
}

// ===========================================================================
extern "C" void kernel_launch(void* const* d_in, const int* in_sizes, int n_in,
                              void* d_out, int out_size, void* d_ws, size_t ws_size,
                              hipStream_t stream)
{
    const float* x    = (const float*)d_in[0];   // [B,S,D]
    const float* w    = (const float*)d_in[1];   // [D,3D]
    const float* bias = (const float*)d_in[2];   // [3D]
    float*       out  = (float*)d_out;           // [B,S,D]

    const size_t head_elems = (size_t)BDIM * HDIM * SDIM * HD;  // 6,291,456
    __bf16* qws = (__bf16*)d_ws;
    __bf16* kws = qws + head_elems;
    __bf16* vws = kws + head_elems;               // total 37.75 MB of d_ws

    // QKV GEMM: 64 M-tiles x 36 N-tiles
    qkv_gemm_kernel<<<dim3(64 * 36), 256, 0, stream>>>(x, w, bias, qws, kws, vws);
    // Flash attention: 32 q-tiles x 48 heads
    flash_attn_kernel<<<dim3(SDIM / 64, BDIM * HDIM), 128, 0, stream>>>(qws, kws, vws, out);
}
